// ModulatedConv2d_65773129171612
// MI455X (gfx1250) — matrix operand order repaired and checked by reading, not compiled, not run on hardware
//
#include <hip/hip_runtime.h>

// ---------------- Types ----------------
typedef unsigned short u16;
typedef unsigned int   u32;
typedef __attribute__((ext_vector_type(16))) __bf16 v16bf;
typedef __attribute__((ext_vector_type(8)))  float  v8f;

// async-LDS builtin pointer types: int vector_size(16) in AS1 (global) / AS3 (LDS)
typedef int avi4 __attribute__((vector_size(16)));
typedef __attribute__((address_space(1))) avi4* gvi4_p;
typedef __attribute__((address_space(3))) avi4* lvi4_p;

#define B_    8
#define CIN   512
#define COUT  512
#define KK    3
#define HH    64
#define WW    64
#define SDIM  512
#define KTOT  (CIN * KK * KK)   // 4608
#define NPIX  (HH * WW)         // 4096

#if defined(__gfx1250__) && __has_builtin(__builtin_amdgcn_global_load_async_to_lds_b128) && __has_builtin(__builtin_amdgcn_s_wait_asynccnt)
#define USE_ASYNC_LDS 1
#else
#define USE_ASYNC_LDS 0
#endif

// round-to-nearest-even f32 -> bf16 (bit pattern)
__device__ __forceinline__ u16 f2bf(float f) {
    unsigned u = __float_as_uint(f);
    unsigned r = u + 0x7FFFu + ((u >> 16) & 1u);
    return (u16)(r >> 16);
}

// load a v16bf WMMA fragment as two 16-byte LDS chunks
__device__ __forceinline__ v16bf load_frag(const u16* base, int off0, int off1) {
    union { uint4 q[2]; v16bf v; } f;
    f.q[0] = *(const uint4*)(base + off0);
    f.q[1] = *(const uint4*)(base + off1);
    return f.v;
}

// ---------------- Kernel 1: s = style @ (mod_w * mod_scale)^T + mod_b ----------------
__global__ void k_modulate(const float* __restrict__ style,
                           const float* __restrict__ mod_w,
                           const float* __restrict__ mod_b,
                           float* __restrict__ s) {
    int idx = blockIdx.x * blockDim.x + threadIdx.x;   // 0 .. B*CIN-1
    if (idx >= B_ * CIN) return;
    int b = idx >> 9;          // /512
    int cin = idx & 511;
    const float mod_scale = rsqrtf((float)SDIM);
    float acc = mod_b[cin];
    const float* st = style + (size_t)b * SDIM;
    const float* mw = mod_w + (size_t)cin * SDIM;
    for (int j = 0; j < SDIM; ++j) acc += st[j] * mw[j] * mod_scale;
    s[idx] = acc;
}

// ---------------- Kernel 2: demod[b][cout] = rsqrt(sum (cs*w*s)^2 + eps) ----------------
__global__ void k_demod(const float* __restrict__ weight,
                        const float* __restrict__ s,
                        float* __restrict__ demod) {
    int b = blockIdx.x >> 9;
    int cout = blockIdx.x & 511;
    int tid = threadIdx.x;
    const float conv_scale = rsqrtf((float)KTOT);
    const float* wrow = weight + (size_t)cout * KTOT;
    const float* srow = s + (size_t)b * CIN;
    float sum = 0.0f;
    for (int k = tid; k < KTOT; k += 256) {
        float wv = conv_scale * wrow[k] * srow[k / 9];
        sum += wv * wv;
    }
    __shared__ float red[256];
    red[tid] = sum;
    __syncthreads();
    for (int off = 128; off > 0; off >>= 1) {
        if (tid < off) red[tid] += red[tid + off];
        __syncthreads();
    }
    if (tid == 0) demod[blockIdx.x] = rsqrtf(red[0] + 1e-8f);
}

// ---------------- Kernel 3: pack bf16 modulated weights [b][rr][cout][cin] ----------------
// rr = ky*3+kx tap index; K-dim (cin) contiguous per (b, rr, cout) row.
__global__ void k_pack(const float* __restrict__ weight,
                       const float* __restrict__ s,
                       const float* __restrict__ demod,
                       u16* __restrict__ wmod) {
    size_t idx = (size_t)blockIdx.x * blockDim.x + threadIdx.x;
    if (idx >= (size_t)B_ * 9 * COUT * CIN) return;
    int cin  = (int)(idx & 511);
    size_t t = idx >> 9;
    int cout = (int)(t & 511);
    t >>= 9;                       // b*9 + rr
    int rr = (int)(t % 9);
    int b  = (int)(t / 9);
    const float conv_scale = rsqrtf((float)KTOT);
    float v = conv_scale * weight[(size_t)cout * KTOT + (size_t)cin * 9 + rr]
            * s[(size_t)b * CIN + cin] * demod[(size_t)b * COUT + cout];
    wmod[idx] = f2bf(v);
}

// ---------------- Kernel 4: implicit-GEMM conv with bf16 WMMA ----------------
// grid: (32 n-tiles, 4 m-tiles, 8 batch), block: 256 threads (8 waves, 4x2 wave grid)
// Output tile 128x128. Outer loop over 9 taps (rr uniform), inner over cin in steps of 32.
#define APITCH 40   // 32 k + 8 pad (u16 units) -> 80B pitch, conflict-free b128 reads

__global__ void __launch_bounds__(256) k_conv(const float* __restrict__ x,
                                              const u16* __restrict__ wmod,
                                              float* __restrict__ out) {
    __shared__ u16 Atile[128 * APITCH];   // [m][k]
    __shared__ u16 Btile[128 * APITCH];   // [n][k]  (K-contiguous per output column)

    const int tid  = threadIdx.x;
    const int lane = tid & 31;
    const int wave = tid >> 5;            // 0..7
    const int wm   = wave & 3;            // 4 waves along M
    const int wn   = wave >> 2;           // 2 waves along N
    const int b     = blockIdx.z;
    const int mbase = blockIdx.y * 128;   // cout base
    const int ntile = blockIdx.x;         // covers image rows 2*ntile, 2*ntile+1
    const int nbase = ntile * 128;

    const float* xb = x + (size_t)b * CIN * NPIX;

    v8f acc[2][4];
    #pragma unroll
    for (int mi = 0; mi < 2; ++mi)
        #pragma unroll
        for (int ni = 0; ni < 4; ++ni) acc[mi][ni] = 0.0f;

    // ---- A-tile fill indices: thread -> (row, half)
    const int arow  = tid >> 1;
    const int ahalf = tid & 1;
    // ---- B-tile fill indices: thread -> (even k-row pair, column segment of 8)
    const int rp   = (tid >> 4) * 2;           // even k-row: 0,2,..,30
    const int cs8  = tid & 15;                 // 0..15 column segments
    const int ytile2 = ntile * 2;
    const int ytid   = ytile2 + (cs8 >> 3);    // output row handled by this thread
    const int xs0    = (cs8 & 7) * 8;          // first output column of segment

    // per-lane fragment offsets (u16 units)
    const int aoff0 = (lane < 16) ? 0 : 8;    // A: K 0-7 / 8-15
    const int aoff1 = (lane < 16) ? 16 : 24;  // A: K 16-23 / 24-31
    const int boff0 = (lane < 16) ? 0 : 8;    // B: K 0-7 / 8-15
    const int boff1 = (lane < 16) ? 16 : 24;  // B: K 16-23 / 24-31
    const int l16 = lane & 15;

    for (int rr = 0; rr < 9; ++rr) {          // tap index (uniform)
        const int ky = rr / 3;                // scalar, hoisted by compiler
        const int kx = rr - ky * 3;
        // vertical coordinate for this tap: clamp + validity (branch-free)
        const int iy  = ytid + ky - 1;
        const bool vy = (iy >= 0) && (iy < HH);
        const int iyc = iy < 0 ? 0 : (iy > HH - 1 ? HH - 1 : iy);
        const int xb0 = xs0 + kx - 1;         // first input column (may be -1 or 64)
        // weight rows for this (b, rr): [cout][cin] contiguous in cin
        const u16* wb = wmod + ((size_t)(b * 9 + rr) * COUT + mbase) * CIN;
        const float* xrow0 = xb + (size_t)rp * NPIX + iyc * WW;  // + cin0*NPIX below

        for (int cin0 = 0; cin0 < CIN; cin0 += 32) {
            __syncthreads();

            // ---- load A tile: 128 rows x 32 cin of bf16 weights (contiguous)
            {
                const u16* src = wb + (size_t)arow * CIN + cin0 + ahalf * 16;
                u16* dst = &Atile[arow * APITCH + ahalf * 16];
#if USE_ASYNC_LDS
                __builtin_amdgcn_global_load_async_to_lds_b128(
                    (gvi4_p)src, (lvi4_p)dst, 0, 0);
                __builtin_amdgcn_global_load_async_to_lds_b128(
                    (gvi4_p)src, (lvi4_p)dst, 16, 0);
#else
                ((uint4*)dst)[0] = ((const uint4*)src)[0];
                ((uint4*)dst)[1] = ((const uint4*)src)[1];
#endif
            }

            // ---- build B tile: k-rows rp, rp+1 (two cin planes), 8 cols, b32-packed
            {
                const float* rowA = xrow0 + (size_t)cin0 * NPIX;   // cin = cin0+rp
                const float* rowB = rowA + NPIX;                   // cin = cin0+rp+1
                #pragma unroll
                for (int j = 0; j < 8; ++j) {
                    int ix = xb0 + j;
                    bool vx = (ix >= 0) && (ix < WW);
                    int ixc = ix < 0 ? 0 : (ix > WW - 1 ? WW - 1 : ix);
                    // loads hoisted out of the select: address is always valid
                    // (clamped), so these lower to plain loads + v_cndmask, no exec
                    float ta = rowA[ixc];
                    float tb = rowB[ixc];
                    bool ok = vy && vx;
                    float va = ok ? ta : 0.0f;
                    float vb = ok ? tb : 0.0f;
                    u32 pk = (u32)f2bf(va) | ((u32)f2bf(vb) << 16);
                    *(u32*)&Btile[(xs0 + j) * APITCH + rp] = pk;  // rp even -> aligned
                }
            }

#if USE_ASYNC_LDS
            __builtin_amdgcn_s_wait_asynccnt(0);   // A-tile LDS writes visible
#endif
            __syncthreads();

            // ---- fragment loads + 8 WMMAs
            v16bf afrag[2], bfrag[4];
            #pragma unroll
            for (int mi = 0; mi < 2; ++mi) {
                const u16* ab = &Atile[(wm * 32 + mi * 16 + l16) * APITCH];
                afrag[mi] = load_frag(ab, aoff0, aoff1);
            }
            #pragma unroll
            for (int ni = 0; ni < 4; ++ni) {
                const u16* bb = &Btile[(wn * 64 + ni * 16 + l16) * APITCH];
                bfrag[ni] = load_frag(bb, boff0, boff1);
            }
            #pragma unroll
            for (int mi = 0; mi < 2; ++mi)
                #pragma unroll
                for (int ni = 0; ni < 4; ++ni)
                    acc[mi][ni] = __builtin_amdgcn_wmma_f32_16x16x32_bf16(
                        false, afrag[mi], false, bfrag[ni],
                        (short)0, acc[mi][ni], false, false);
        }
    }

    // ---- store: C/D layout — VGPR i: lanes 0-15 => M=i, lanes 16-31 => M=i+8; N=lane%16
    const int mrow0 = mbase + wm * 32 + ((lane < 16) ? 0 : 8);
    const int col0  = nbase + wn * 64 + l16;
    #pragma unroll
    for (int mi = 0; mi < 2; ++mi) {
        #pragma unroll
        for (int ni = 0; ni < 4; ++ni) {
            #pragma unroll
            for (int i = 0; i < 8; ++i) {
                int row = mrow0 + mi * 16 + i;
                int col = col0 + ni * 16;
                out[((size_t)b * COUT + row) * NPIX + col] = acc[mi][ni][i];
            }
        }
    }
}

// ---------------- Launch ----------------
extern "C" void kernel_launch(void* const* d_in, const int* in_sizes, int n_in,
                              void* d_out, int out_size, void* d_ws, size_t ws_size,
                              hipStream_t stream) {
    const float* x      = (const float*)d_in[0];
    const float* style  = (const float*)d_in[1];
    const float* weight = (const float*)d_in[2];
    const float* mod_w  = (const float*)d_in[3];
    const float* mod_b  = (const float*)d_in[4];
    float* out = (float*)d_out;

    float* s     = (float*)d_ws;             // B*CIN  = 4096 f32
    float* demod = s + B_ * CIN;             // B*COUT = 4096 f32
    u16*   wmod  = (u16*)(demod + B_ * COUT);// B*9*COUT*CIN bf16 = 37.75 MB

    // 1) style modulation
    k_modulate<<<(B_ * CIN + 255) / 256, 256, 0, stream>>>(style, mod_w, mod_b, s);
    // 2) demodulation factors
    k_demod<<<B_ * COUT, 256, 0, stream>>>(weight, s, demod);
    // 3) pack bf16 modulated weights, tap-major layout [b][rr][cout][cin]
    {
        size_t total = (size_t)B_ * 9 * COUT * CIN;
        k_pack<<<(unsigned)((total + 255) / 256), 256, 0, stream>>>(weight, s, demod, wmod);
    }
    // 4) implicit-GEMM conv, WMMA bf16 -> f32
    k_conv<<<dim3(NPIX / 128, COUT / 128, B_), 256, 0, stream>>>(x, wmod, out);
}
